// DGCNN_52321291600315
// MI455X (gfx1250) — matrix operand order repaired and checked by reading, compiled, not run on hardware
//
#include <hip/hip_runtime.h>
#include <hip/hip_bf16.h>

// ---------------- problem constants ----------------
constexpr int  BG    = 512;          // graphs
constexpr int  PP    = 64;           // nodes per graph
constexpr int  KPOOL = 30;           // sort-pool k
constexpr int  NN    = BG * PP;      // 32768 nodes
constexpr long EE    = 524288;       // edges
constexpr int  FF    = 128;          // input features
constexpr int  HH    = 256;          // hidden
constexpr int  LOUT  = 27;           // conv1d output length
constexpr int  NCLS  = 10;

// ---------------- WMMA types ----------------
typedef __attribute__((ext_vector_type(16))) __bf16 v16bf;
typedef __attribute__((ext_vector_type(8)))  float  v8f;

union FragU { v16bf v; unsigned u[8]; uint4 q[2]; };

__device__ __forceinline__ unsigned short f2bf(float f) {
    unsigned u = __float_as_uint(f);
    u = (u + 0x7FFFu + ((u >> 16) & 1u)) >> 16;   // RNE
    return (unsigned short)u;
}
__device__ __forceinline__ float bf2f(unsigned short h) {
    return __uint_as_float((unsigned)h << 16);
}

// A fragment: bf16 row-major; ISA 16x32 A layout.
// lanes 0-15: VGPR j<4 -> K=2j,2j+1 ; j>=4 -> K=16+2(j-4)..  (+8 for lanes 16-31)
// => two contiguous 16B groups at (row*lda + kk + hs*8) and +16 elements.
__device__ __forceinline__ v16bf load_a_frag(const unsigned short* __restrict__ A, long lda,
                                             int row, int kk, int hs) {
    FragU f;
    const unsigned short* base = A + (long)row * lda + kk + hs * 8;
    f.q[0] = *(const uint4*)(base);
    f.q[1] = *(const uint4*)(base + 16);
    return f.v;
}
// B fragment: Wt[HoPad x K] bf16 row-major (= col-major of W).
// VGPR j<4 -> K=2j,2j+1 ; j>=4 -> K=16+2(j-4).. (+8 for lanes 16-31): two 16B runs.
__device__ __forceinline__ v16bf load_b_frag(const unsigned short* __restrict__ Wt, int K,
                                             int col, int kk, int hs) {
    FragU f;
    const unsigned short* base = Wt + (long)col * K + kk + hs * 8;
    f.q[0] = *(const uint4*)(base);
    f.q[1] = *(const uint4*)(base + 16);
    return f.v;
}

// ---------------- GEMM kernels ----------------
// Block: 256 threads = 8 waves; each wave computes a 16x32 C tile (two 16x16 WMMA tiles,
// A fragment reused) -> block tile 128 rows x 32 cols.
// SAGE layer: C = relu(A1@W1 + A2@W2 + bias), bf16 output.
__global__ void k_gemm_dual(const unsigned short* __restrict__ A1,
                            const unsigned short* __restrict__ A2,
                            int K, const unsigned short* __restrict__ W1t,
                            const unsigned short* __restrict__ W2t,
                            const float* __restrict__ bias,
                            unsigned short* __restrict__ C, int ldc) {
    int lane = threadIdx.x & 31, w = threadIdx.x >> 5;
    int hs = lane >> 4, mr = lane & 15;
    int m_base = blockIdx.x * 128 + w * 16;
    int n_base = blockIdx.y * 32;
    v8f c0 = {0.f, 0.f, 0.f, 0.f, 0.f, 0.f, 0.f, 0.f};
    v8f c1 = {0.f, 0.f, 0.f, 0.f, 0.f, 0.f, 0.f, 0.f};
    for (int kk = 0; kk < K; kk += 32) {
        v16bf a1 = load_a_frag(A1, K, m_base + mr, kk, hs);
        v16bf a2 = load_a_frag(A2, K, m_base + mr, kk, hs);
        v16bf b10 = load_b_frag(W1t, K, n_base + mr, kk, hs);
        v16bf b11 = load_b_frag(W1t, K, n_base + 16 + mr, kk, hs);
        v16bf b20 = load_b_frag(W2t, K, n_base + mr, kk, hs);
        v16bf b21 = load_b_frag(W2t, K, n_base + 16 + mr, kk, hs);
        c0 = __builtin_amdgcn_wmma_f32_16x16x32_bf16(false, a1, false, b10, (short)0, c0, false, false);
        c1 = __builtin_amdgcn_wmma_f32_16x16x32_bf16(false, a1, false, b11, (short)0, c1, false, false);
        c0 = __builtin_amdgcn_wmma_f32_16x16x32_bf16(false, a2, false, b20, (short)0, c0, false, false);
        c1 = __builtin_amdgcn_wmma_f32_16x16x32_bf16(false, a2, false, b21, (short)0, c1, false, false);
    }
    int col0 = n_base + mr, col1 = col0 + 16;
    float bv0 = bias[col0], bv1 = bias[col1];
#pragma unroll
    for (int r = 0; r < 8; ++r) {
        int row = m_base + r + hs * 8;
        float v0 = c0[r] + bv0; v0 = v0 > 0.f ? v0 : 0.f;
        float v1 = c1[r] + bv1; v1 = v1 > 0.f ? v1 : 0.f;
        C[(long)row * ldc + col0] = f2bf(v0);
        C[(long)row * ldc + col1] = f2bf(v1);
    }
}

// General single GEMM: z-sliced A (conv1d), strided C store, relu always.
// BF16OUT selects bf16 (intermediate) vs fp32 (final logits) store.
template <bool BF16OUT>
__global__ void k_gemm1(const unsigned short* __restrict__ A, long lda, long a_z, int K,
                        const unsigned short* __restrict__ Wt,
                        const float* __restrict__ bias, int n_valid,
                        void* __restrict__ Cout, long c_row, int c_col, long c_z) {
    int lane = threadIdx.x & 31, w = threadIdx.x >> 5;
    int hs = lane >> 4, mr = lane & 15;
    int m_base = blockIdx.x * 128 + w * 16;
    int n_base = blockIdx.y * 32;
    const unsigned short* __restrict__ Ab = A + (long)blockIdx.z * a_z;
    v8f c0 = {0.f, 0.f, 0.f, 0.f, 0.f, 0.f, 0.f, 0.f};
    v8f c1 = {0.f, 0.f, 0.f, 0.f, 0.f, 0.f, 0.f, 0.f};
    for (int kk = 0; kk < K; kk += 32) {
        v16bf a  = load_a_frag(Ab, lda, m_base + mr, kk, hs);
        v16bf b0 = load_b_frag(Wt, K, n_base + mr, kk, hs);
        v16bf b1 = load_b_frag(Wt, K, n_base + 16 + mr, kk, hs);
        c0 = __builtin_amdgcn_wmma_f32_16x16x32_bf16(false, a, false, b0, (short)0, c0, false, false);
        c1 = __builtin_amdgcn_wmma_f32_16x16x32_bf16(false, a, false, b1, (short)0, c1, false, false);
    }
    int col0 = n_base + mr, col1 = col0 + 16;
    float bv0 = (col0 < n_valid) ? bias[col0] : 0.f;
    float bv1 = (col1 < n_valid) ? bias[col1] : 0.f;
#pragma unroll
    for (int r = 0; r < 8; ++r) {
        int row = m_base + r + hs * 8;
        float v0 = c0[r] + bv0; v0 = v0 > 0.f ? v0 : 0.f;
        float v1 = c1[r] + bv1; v1 = v1 > 0.f ? v1 : 0.f;
        long i0 = (long)row * c_row + (long)col0 * c_col + (long)blockIdx.z * c_z;
        long i1 = (long)row * c_row + (long)col1 * c_col + (long)blockIdx.z * c_z;
        if (BF16OUT) {
            if (col0 < n_valid) ((unsigned short*)Cout)[i0] = f2bf(v0);
            if (col1 < n_valid) ((unsigned short*)Cout)[i1] = f2bf(v1);
        } else {
            if (col0 < n_valid) ((float*)Cout)[i0] = v0;
            if (col1 < n_valid) ((float*)Cout)[i1] = v1;
        }
    }
}

// ---------------- elementwise / graph kernels ----------------
__global__ void k_zero(float* __restrict__ p, long n) {
    long i = (long)blockIdx.x * blockDim.x + threadIdx.x;
    if (i < n) p[i] = 0.f;
}
__global__ void k_nan0(const float* __restrict__ x, unsigned short* __restrict__ o, long n) {
    long i = (long)blockIdx.x * blockDim.x + threadIdx.x;
    if (i < n) { float v = x[i]; o[i] = f2bf((v != v) ? 0.f : v); }
}
__global__ void k_degcnt(const int* __restrict__ dst, float* __restrict__ deg) {
    long e = (long)blockIdx.x * blockDim.x + threadIdx.x;
    if (e < EE) atomicAdd(&deg[dst[e]], 1.0f);
}
__global__ void k_invdeg(float* __restrict__ deg) {
    long i = (long)blockIdx.x * blockDim.x + threadIdx.x;
    if (i < NN) deg[i] = 1.0f / fmaxf(deg[i], 1.0f);
}
// scatter-add: one thread per (edge, 4-feature chunk); bf16 gather -> fp32 atomics.
__global__ void k_agg(const int* __restrict__ src, const int* __restrict__ dst,
                      const unsigned short* __restrict__ h, float* __restrict__ agg,
                      int K, int cshift) {
    long idx = (long)blockIdx.x * blockDim.x + threadIdx.x;
    long total = EE << cshift;
    if (idx >= total) return;
    long e = idx >> cshift;
    int c = (int)(idx & ((1 << cshift) - 1)) << 2;
    uint2 d = *(const uint2*)(h + (long)src[e] * K + c);
    float* a = agg + (long)dst[e] * K + c;
    atomicAdd(a + 0, bf2f((unsigned short)(d.x & 0xFFFFu)));
    atomicAdd(a + 1, bf2f((unsigned short)(d.x >> 16)));
    atomicAdd(a + 2, bf2f((unsigned short)(d.y & 0xFFFFu)));
    atomicAdd(a + 3, bf2f((unsigned short)(d.y >> 16)));
}
// agg *= inv_deg, convert to bf16 GEMM operand
__global__ void k_scale(const float* __restrict__ agg, const float* __restrict__ invdeg,
                        unsigned short* __restrict__ aggb, long n, int kshift) {
    long i = (long)blockIdx.x * blockDim.x + threadIdx.x;
    if (i < n) aggb[i] = f2bf(agg[i] * invdeg[i >> kshift]);
}

// ---------------- sort-pool: per-graph stable top-30 by channel 255 ----------------
__global__ void k_sortpool(const unsigned short* __restrict__ h,
                           unsigned short* __restrict__ topk) {
    __shared__ float key[PP];
    __shared__ int order[KPOOL];
    int g = blockIdx.x, t = threadIdx.x;
    const unsigned short* hg = h + (long)g * PP * HH;
    key[t] = bf2f(hg[t * HH + (HH - 1)]);
    __syncthreads();
    float kt = key[t];
    int rank = 0;
#pragma unroll 8
    for (int j = 0; j < PP; ++j) {
        float kj = key[j];
        rank += (kj > kt) || (kj == kt && j < t);
    }
    if (rank < KPOOL) order[rank] = t;
    __syncthreads();
    for (int idx = t; idx < KPOOL * HH; idx += PP) {
        int r = idx >> 8, c = idx & (HH - 1);
        topk[(long)g * KPOOL * HH + idx] = hg[(long)order[r] * HH + c];
    }
}

// ---------------- weight packing (fp32 [KxHo] -> bf16 Wt[HoPad x K], transposed) ----------------
__global__ void k_pack_wt(const float* __restrict__ W, unsigned short* __restrict__ Wt,
                          int K, int Ho, long total) {
    long i = (long)blockIdx.x * blockDim.x + threadIdx.x;
    if (i >= total) return;
    int o = (int)(i / K), k = (int)(i % K);
    float v = (o < Ho) ? W[(long)k * Ho + o] : 0.f;
    Wt[i] = f2bf(v);
}
// conv1d_w (O,I,T)=(256,256,4) row-major -> Wt[o][t*256+i]
__global__ void k_pack_conv(const float* __restrict__ W, unsigned short* __restrict__ Wt) {
    long i = (long)blockIdx.x * blockDim.x + threadIdx.x;
    if (i >= (long)HH * HH * 4) return;
    int o = (int)(i >> 10);
    int r = (int)(i & 1023);
    int t = r >> 8, ii = r & 255;
    Wt[i] = f2bf(W[(long)o * 1024 + ii * 4 + t]);
}

// ---------------- launcher ----------------
static inline int cdiv(long a, long b) { return (int)((a + b - 1) / b); }

extern "C" void kernel_launch(void* const* d_in, const int* in_sizes, int n_in,
                              void* d_out, int out_size, void* d_ws, size_t ws_size,
                              hipStream_t stream) {
    (void)in_sizes; (void)n_in; (void)out_size; (void)ws_size;
    const float* x    = (const float*)d_in[0];
    const int*   ei   = (const int*)d_in[1];
    const int*   src  = ei;
    const int*   dst  = ei + EE;
    const float* swl[4] = {(const float*)d_in[3], (const float*)d_in[6],
                           (const float*)d_in[9], (const float*)d_in[12]};
    const float* swr[4] = {(const float*)d_in[4], (const float*)d_in[7],
                           (const float*)d_in[10], (const float*)d_in[13]};
    const float* sb[4]  = {(const float*)d_in[5], (const float*)d_in[8],
                           (const float*)d_in[11], (const float*)d_in[14]};
    const float* convw = (const float*)d_in[15];
    const float* convb = (const float*)d_in[16];
    const float* l1w = (const float*)d_in[17]; const float* l1b = (const float*)d_in[18];
    const float* l2w = (const float*)d_in[19]; const float* l2b = (const float*)d_in[20];
    const float* ow  = (const float*)d_in[21]; const float* ob  = (const float*)d_in[22];
    float* out = (float*)d_out;

    // ---- workspace layout (bf16 feature maps) ----
    char* ws = (char*)d_ws;
    constexpr size_t SZ_HB = (size_t)NN * HH * 2;   // 16 MB bf16 node features
    unsigned short* hA   = (unsigned short*)(ws + 0);
    unsigned short* hB   = (unsigned short*)(ws + SZ_HB);
    float*          agg  = (float*)(ws + 2 * SZ_HB);               // fp32 accum, 32 MB
    unsigned short* aggb = (unsigned short*)(ws + 4 * SZ_HB);      // bf16 operand, 16 MB
    unsigned short* topk = (unsigned short*)agg;                   // reuse after SAGE (7.9 MB)
    unsigned short* yconv = (unsigned short*)(ws + 2 * SZ_HB + (8ul << 20)); // 7 MB
    float* deg  = (float*)(ws + 5 * SZ_HB);
    unsigned short* l1o = (unsigned short*)(ws + 5 * SZ_HB + (size_t)NN * 4);
    unsigned short* l2o = l1o + (size_t)BG * HH;
    unsigned short* wb  = l2o + (size_t)BG * 128;
    unsigned short *wlT[4], *wrT[4];
    size_t wo = 0;
    wlT[0] = wb + wo; wo += (size_t)HH * FF;
    wrT[0] = wb + wo; wo += (size_t)HH * FF;
    for (int l = 1; l < 4; ++l) {
        wlT[l] = wb + wo; wo += (size_t)HH * HH;
        wrT[l] = wb + wo; wo += (size_t)HH * HH;
    }
    unsigned short* convT = wb + wo; wo += (size_t)HH * 1024;
    unsigned short* l1T   = wb + wo; wo += (size_t)HH * (HH * LOUT);
    unsigned short* l2T   = wb + wo; wo += (size_t)128 * HH;
    unsigned short* outT  = wb + wo; wo += (size_t)32 * 128;       // pad 10 -> 32 cols

    const int TB = 256;

    // ---- pack all weights ----
    k_pack_wt<<<cdiv((long)HH * FF, TB), TB, 0, stream>>>(swl[0], wlT[0], FF, HH, (long)HH * FF);
    k_pack_wt<<<cdiv((long)HH * FF, TB), TB, 0, stream>>>(swr[0], wrT[0], FF, HH, (long)HH * FF);
    for (int l = 1; l < 4; ++l) {
        k_pack_wt<<<cdiv((long)HH * HH, TB), TB, 0, stream>>>(swl[l], wlT[l], HH, HH, (long)HH * HH);
        k_pack_wt<<<cdiv((long)HH * HH, TB), TB, 0, stream>>>(swr[l], wrT[l], HH, HH, (long)HH * HH);
    }
    k_pack_conv<<<cdiv((long)HH * 1024, TB), TB, 0, stream>>>(convw, convT);
    k_pack_wt<<<cdiv((long)HH * HH * LOUT, TB), TB, 0, stream>>>(l1w, l1T, HH * LOUT, HH, (long)HH * HH * LOUT);
    k_pack_wt<<<cdiv((long)128 * HH, TB), TB, 0, stream>>>(l2w, l2T, HH, 128, (long)128 * HH);
    k_pack_wt<<<cdiv((long)32 * 128, TB), TB, 0, stream>>>(ow, outT, 128, NCLS, (long)32 * 128);

    // ---- degrees ----
    k_zero<<<cdiv(NN, TB), TB, 0, stream>>>(deg, NN);
    k_degcnt<<<cdiv(EE, TB), TB, 0, stream>>>(dst, deg);
    k_invdeg<<<cdiv(NN, TB), TB, 0, stream>>>(deg);

    // ---- h0 = bf16(nan_to_num(x)) ----
    k_nan0<<<cdiv((long)NN * FF, TB), TB, 0, stream>>>(x, hA, (long)NN * FF);

    // ---- 4 SAGE layers (ping-pong hA <-> hB) ----
    unsigned short* hcur = hA;
    unsigned short* hnext = hB;
    for (int l = 0; l < 4; ++l) {
        int K = (l == 0) ? FF : HH;
        int cshift = (K == 128) ? 5 : 6;   // log2(K/4)
        int kshift = (K == 128) ? 7 : 8;   // log2(K)
        long nk = (long)NN * K;
        k_zero<<<cdiv(nk, TB), TB, 0, stream>>>(agg, nk);
        k_agg<<<cdiv(EE << cshift, TB), TB, 0, stream>>>(src, dst, hcur, agg, K, cshift);
        k_scale<<<cdiv(nk, TB), TB, 0, stream>>>(agg, deg, aggb, nk, kshift);
        dim3 g(NN / 128, HH / 32, 1);
        k_gemm_dual<<<g, TB, 0, stream>>>(aggb, hcur, K, wlT[l], wrT[l], sb[l], hnext, HH);
        unsigned short* t = hcur; hcur = hnext; hnext = t;
    }
    // final features in hcur (== hA after 4 swaps)

    // ---- sort-pool -> topk [512, 30, 256] bf16 ----
    k_sortpool<<<BG, PP, 0, stream>>>(hcur, topk);

    // ---- conv1d as 27 shifted GEMMs: [512x1024]@[1024x256] -> yconv[b, o*27+l] ----
    {
        dim3 g(BG / 128, HH / 32, LOUT);
        k_gemm1<true><<<g, TB, 0, stream>>>(topk, (long)KPOOL * HH, (long)HH, 4 * HH,
                                            convT, convb, HH, yconv, (long)HH * LOUT, LOUT, 1);
    }
    // ---- lin1: [512 x 6912] @ [6912 x 256] ----
    {
        dim3 g(BG / 128, HH / 32, 1);
        k_gemm1<true><<<g, TB, 0, stream>>>(yconv, (long)HH * LOUT, 0, HH * LOUT, l1T, l1b, HH,
                                            l1o, HH, 1, 0);
    }
    // ---- lin2: [512 x 256] @ [256 x 128] ----
    {
        dim3 g(BG / 128, 128 / 32, 1);
        k_gemm1<true><<<g, TB, 0, stream>>>(l1o, HH, 0, HH, l2T, l2b, 128, l2o, 128, 1, 0);
    }
    // ---- out: [512 x 128] @ [128 x 10] (padded to 32 cols), fp32 store to d_out ----
    {
        dim3 g(BG / 128, 1, 1);
        k_gemm1<false><<<g, TB, 0, stream>>>(l2o, 128, 0, 128, outT, ob, NCLS, out, NCLS, 1, 0);
    }
}
